// Seq2SeqTagger_85169201479938
// MI455X (gfx1250) — matrix operand order repaired and checked by reading, not compiled, not run on hardware
//
#include <hip/hip_runtime.h>
#include <hip/hip_bf16.h>

// ---------------------------------------------------------------------------
// Seq2Seq LSTM tagger on gfx1250 (MI455X), bf16 WMMA (v_wmma_f32_16x16x32_bf16)
//   V=50257 E=512 H=1024 K=64 B=64 T=512
// Strategy:
//   - once per launch: transpose/convert weights f32 -> n-major bf16 (Wt[n,k])
//   - per timestep: fused kernel computes gates = [x,h] @ W + b via WMMA and
//     performs the LSTM cell update in registers (i/f/g/o never hit memory)
//   - h ping-pongs between two bf16 buffers (cross-step RAW handled by
//     stream-ordered kernel launches)
//   - decoder adds a per-step 64x64 WMMA FC writing f32 logits to d_out
// ---------------------------------------------------------------------------

typedef __attribute__((ext_vector_type(16))) __bf16 v16bf;
typedef __attribute__((ext_vector_type(8)))  __bf16 v8bf;
typedef __attribute__((ext_vector_type(8)))  float  v8f;

#define H_DIM   1024
#define B_DIM   64
#define T_LEN   512
#define E_DIM   512
#define K_TAGS  64

// ---- WMMA fragment loaders (wave32) ---------------------------------------
// A-matrix 16x32 bf16, row-major source (ld elements per row).
// Lane l<16 : row=l,    K = k0+0..15
// Lane l>=16: row=l-16, K = k0+16..31
__device__ __forceinline__ v16bf load_frag_a(const __bf16* base, int ld,
                                             int row0, int k0) {
    const int lane = threadIdx.x & 31;
    const int r    = row0 + (lane & 15);
    const int kb   = k0 + ((lane >> 4) << 4);
    return *(const v16bf*)(base + (size_t)r * ld + kb);
}

// B-matrix 32x16 bf16 from n-major Wt (Wt[n,k], ld = K-extent).
// Lane l<16 : col=l,    K = k0+0..15 ; lane l>=16: col=l-16, K = k0+16..31
__device__ __forceinline__ v16bf load_frag_b(const __bf16* base, int ld,
                                             int n0, int k0) {
    const int lane = threadIdx.x & 31;
    const int n    = n0 + (lane & 15);
    const int k    = k0 + ((lane >> 4) << 4);
    return *(const v16bf*)(base + (size_t)n * ld + k);
}

__device__ __forceinline__ float sigm(float x) {
    return 1.0f / (1.0f + __expf(-x));
}

// ---- weight convert: W[K,N] f32 row-major -> Wt[N,K] bf16 -----------------
__global__ void wconvert_kernel(const float* __restrict__ W,
                                __bf16* __restrict__ Wt, int K, int N) {
    int idx = blockIdx.x * blockDim.x + threadIdx.x;
    if (idx >= K * N) return;
    int n = idx / K;
    int k = idx - n * K;
    Wt[(size_t)n * K + k] = (__bf16)W[(size_t)k * N + n];
}

// ---- state init: h (bf16) = 0, c (f32) = 0 --------------------------------
__global__ void init_state_kernel(__bf16* __restrict__ zh,
                                  float* __restrict__ c) {
    int idx = blockIdx.x * blockDim.x + threadIdx.x;
    if (idx >= B_DIM * H_DIM) return;
    zh[idx] = (__bf16)0.0f;
    c[idx]  = 0.0f;
}

// ---- encoder x: zx[b, :] = bf16(emb_table[src[b,t], :]) -------------------
__global__ void gather_emb_kernel(const int* __restrict__ src,
                                  const float* __restrict__ emb,
                                  __bf16* __restrict__ zx, int t) {
    int idx = blockIdx.x * blockDim.x + threadIdx.x;   // B*E = 32768
    if (idx >= B_DIM * E_DIM) return;
    int b = idx >> 9;           // /512
    int e = idx & (E_DIM - 1);
    int tok = src[b * T_LEN + t];
    zx[idx] = (__bf16)emb[(size_t)tok * E_DIM + e];
}

// ---- decoder x: one-hot of trg_in[b,t] ------------------------------------
__global__ void onehot_kernel(const int* __restrict__ trg,
                              __bf16* __restrict__ zx, int t) {
    int idx = blockIdx.x * blockDim.x + threadIdx.x;   // B*K = 4096
    if (idx >= B_DIM * K_TAGS) return;
    int b = idx >> 6;
    int k = idx & (K_TAGS - 1);
    zx[idx] = (__bf16)((trg[b * T_LEN + t] == k) ? 1.0f : 0.0f);
}

// ---- fused LSTM step: gates = [zx|zh] @ Wt^T + b ; cell update ------------
// grid: 32 blocks x 256 threads (8 waves). Block owns 32 h-columns.
// Wave w: m-tile = w>>1 (rows m*16..m*16+15), n-sub = w&1 (16 cols).
// Each wave keeps 4 accumulators (i,f,g,o) for its 16x16 (b,n) patch.
__global__ void lstm_step_kernel(const __bf16* __restrict__ zx, int Kx,
                                 const __bf16* __restrict__ zh_in,
                                 const __bf16* __restrict__ Wt, int ldW,
                                 const float* __restrict__ bias,
                                 float* __restrict__ cbuf,
                                 __bf16* __restrict__ zh_out) {
    const int wave  = threadIdx.x >> 5;
    const int lane  = threadIdx.x & 31;
    const int row0  = (wave >> 1) * 16;                   // batch-row tile
    const int ncol0 = blockIdx.x * 32 + (wave & 1) * 16;  // h column base

    v8f acc[4] = {v8f{}, v8f{}, v8f{}, v8f{}};

    // phase 0: x contribution (K = 0 .. Kx)
    for (int k0 = 0; k0 < Kx; k0 += 32) {
        v16bf a = load_frag_a(zx, Kx, row0, k0);
#pragma unroll
        for (int g = 0; g < 4; ++g) {
            v16bf b = load_frag_b(Wt, ldW, g * H_DIM + ncol0, k0);
            acc[g] = __builtin_amdgcn_wmma_f32_16x16x32_bf16(
                false, a, false, b, (short)0, acc[g], false, false);
        }
    }
    // phase 1: h contribution (K = Kx .. Kx+1024)
    for (int k0 = 0; k0 < H_DIM; k0 += 32) {
        // prefetch next weight chunk (L2 temporal) — global_prefetch path
        __builtin_prefetch(Wt + (size_t)ncol0 * ldW + Kx + k0 + 32, 0, 1);
        v16bf a = load_frag_a(zh_in, H_DIM, row0, k0);
#pragma unroll
        for (int g = 0; g < 4; ++g) {
            v16bf b = load_frag_b(Wt, ldW, g * H_DIM + ncol0, Kx + k0);
            acc[g] = __builtin_amdgcn_wmma_f32_16x16x32_bf16(
                false, a, false, b, (short)0, acc[g], false, false);
        }
    }

    // cell update. C/D layout: lane holds col = lane&15,
    // rows r (lanes 0-15) / r+8 (lanes 16-31) for acc element r.
    const int col   = ncol0 + (lane & 15);
    const int rbase = row0 + ((lane >> 4) << 3);
    const float bi = bias[col];
    const float bf = bias[H_DIM + col];
    const float bg = bias[2 * H_DIM + col];
    const float bo = bias[3 * H_DIM + col];
#pragma unroll
    for (int r = 0; r < 8; ++r) {
        const int row = rbase + r;
        const size_t off = (size_t)row * H_DIM + col;
        float iv = sigm(acc[0][r] + bi);
        float fv = sigm(acc[1][r] + bf);
        float gv = tanhf(acc[2][r] + bg);
        float ov = sigm(acc[3][r] + bo);
        float cn = fv * cbuf[off] + iv * gv;
        cbuf[off]   = cn;
        zh_out[off] = (__bf16)(ov * tanhf(cn));
    }
}

// ---- decoder FC: logits[b,t,:] = h @ fc_W + fc_b (64x64, K=1024) ----------
// 1 block x 256 threads; wave w: m-tile = w>>1, two n-tiles = (w&1)*2 + j
__global__ void fc_step_kernel(const __bf16* __restrict__ zh,
                               const __bf16* __restrict__ fcWt,
                               const float* __restrict__ fcb,
                               float* __restrict__ out, int t) {
    const int wave = threadIdx.x >> 5;
    const int lane = threadIdx.x & 31;
    const int row0 = (wave >> 1) * 16;
    const int nb   = (wave & 1) * 32;

    v8f acc[2] = {v8f{}, v8f{}};
    for (int k0 = 0; k0 < H_DIM; k0 += 32) {
        v16bf a = load_frag_a(zh, H_DIM, row0, k0);
#pragma unroll
        for (int j = 0; j < 2; ++j) {
            v16bf b = load_frag_b(fcWt, H_DIM, nb + j * 16, k0);
            acc[j] = __builtin_amdgcn_wmma_f32_16x16x32_bf16(
                false, a, false, b, (short)0, acc[j], false, false);
        }
    }
#pragma unroll
    for (int j = 0; j < 2; ++j) {
        const int col   = nb + j * 16 + (lane & 15);
        const int rbase = row0 + ((lane >> 4) << 3);
        const float bb  = fcb[col];
#pragma unroll
        for (int r = 0; r < 8; ++r) {
            const int b = rbase + r;
            out[((size_t)b * T_LEN + t) * K_TAGS + col] = acc[j][r] + bb;
        }
    }
}

// ---------------------------------------------------------------------------
extern "C" void kernel_launch(void* const* d_in, const int* in_sizes, int n_in,
                              void* d_out, int out_size, void* d_ws,
                              size_t ws_size, hipStream_t stream) {
    (void)in_sizes; (void)n_in; (void)out_size; (void)ws_size;
    const int*   src  = (const int*)d_in[0];
    const int*   trg  = (const int*)d_in[1];
    const float* emb  = (const float*)d_in[2];
    const float* encW = (const float*)d_in[3];   // [1536, 4096]
    const float* encb = (const float*)d_in[4];
    const float* decW = (const float*)d_in[5];   // [1088, 4096]
    const float* decb = (const float*)d_in[6];
    const float* fcW  = (const float*)d_in[7];   // [1024, 64]
    const float* fcb  = (const float*)d_in[8];
    float* out = (float*)d_out;

    const int KE = E_DIM + H_DIM;    // 1536
    const int KD = K_TAGS + H_DIM;   // 1088

    // workspace layout (bytes)
    char* ws = (char*)d_ws;
    size_t off = 0;
    __bf16* encWt = (__bf16*)(ws + off); off += (size_t)4096 * KE * 2;      // 12.6 MB
    __bf16* decWt = (__bf16*)(ws + off); off += (size_t)4096 * KD * 2;      //  8.9 MB
    __bf16* fcWt  = (__bf16*)(ws + off); off += (size_t)K_TAGS * H_DIM * 2; // 128 KB
    __bf16* zx    = (__bf16*)(ws + off); off += (size_t)B_DIM * E_DIM * 2;  //  64 KB
    __bf16* zhA   = (__bf16*)(ws + off); off += (size_t)B_DIM * H_DIM * 2;  // 128 KB
    __bf16* zhB   = (__bf16*)(ws + off); off += (size_t)B_DIM * H_DIM * 2;  // 128 KB
    float*  cbuf  = (float*)(ws + off);                                     // 256 KB

    // ---- one-time (per launch) weight converts + state init ----
    {
        int n;
        n = KE * 4096;
        wconvert_kernel<<<(n + 255) / 256, 256, 0, stream>>>(encW, encWt, KE, 4096);
        n = KD * 4096;
        wconvert_kernel<<<(n + 255) / 256, 256, 0, stream>>>(decW, decWt, KD, 4096);
        n = H_DIM * K_TAGS;
        wconvert_kernel<<<(n + 255) / 256, 256, 0, stream>>>(fcW, fcWt, H_DIM, K_TAGS);
        n = B_DIM * H_DIM;
        init_state_kernel<<<(n + 255) / 256, 256, 0, stream>>>(zhA, cbuf);
    }

    __bf16* zh[2] = {zhA, zhB};
    int p = 0;

    // ---- encoder: 512 sequential steps ----
    for (int t = 0; t < T_LEN; ++t) {
        gather_emb_kernel<<<(B_DIM * E_DIM + 255) / 256, 256, 0, stream>>>(
            src, emb, zx, t);
        lstm_step_kernel<<<H_DIM / 32, 256, 0, stream>>>(
            zx, E_DIM, zh[p], encWt, KE, encb, cbuf, zh[1 - p]);
        p ^= 1;
    }

    // ---- decoder: 512 sequential steps + FC ----
    for (int t = 0; t < T_LEN; ++t) {
        onehot_kernel<<<(B_DIM * K_TAGS + 255) / 256, 256, 0, stream>>>(
            trg, zx, t);
        lstm_step_kernel<<<H_DIM / 32, 256, 0, stream>>>(
            zx, K_TAGS, zh[p], decWt, KD, decb, cbuf, zh[1 - p]);
        p ^= 1;
        fc_step_kernel<<<1, 256, 0, stream>>>(zh[p], fcWt, fcb, out, t);
    }
}
